// HRM_54391465837136
// MI455X (gfx1250) — matrix operand order repaired and checked by reading, not compiled, not run on hardware
//
#include <hip/hip_runtime.h>
#include <hip/hip_bf16.h>
#include <math.h>

#define C_NODES 10000
#define E_EDGES 160000
#define EWL     170000   /* E + C self loops */
#define D_DIM   256

#ifndef __has_builtin
#define __has_builtin(x) 0
#endif
#if __has_builtin(__builtin_amdgcn_global_load_async_to_lds_b128) && \
    __has_builtin(__builtin_amdgcn_s_wait_asynccnt)
#define HAVE_ASYNC_LDS 1
#else
#define HAVE_ASYNC_LDS 0
#endif

typedef __attribute__((ext_vector_type(16))) __bf16 v16bf;
typedef __attribute__((ext_vector_type(8)))  float  v8f;
typedef __attribute__((ext_vector_type(4)))  int    v4i;

__device__ __forceinline__ unsigned short f2bf(float x) {
    unsigned u = __float_as_uint(x);
    unsigned r = u + 0x7FFFu + ((u >> 16) & 1u);
    return (unsigned short)(r >> 16);
}
__device__ __forceinline__ float geluf(float x) {
    return 0.5f * x * (1.0f + erff(x * 0.70710678118654752f));
}
__device__ __forceinline__ float leakyf(float x) { return x >= 0.f ? x : 0.2f * x; }
__device__ __forceinline__ void atomicMaxF(float* a, float v) {
    if (v >= 0.f) atomicMax((int*)a, __float_as_int(v));
    else          atomicMin((unsigned int*)a, (unsigned int)__float_as_int(v));
}

__device__ __forceinline__ void stage16(const unsigned short* gp, unsigned short* lp) {
#if HAVE_ASYNC_LDS
    __builtin_amdgcn_global_load_async_to_lds_b128(
        (__attribute__((address_space(1))) v4i*)gp,
        (__attribute__((address_space(3))) v4i*)lp, 0, 0);
#else
    *(uint4*)lp = *(const uint4*)gp;
#endif
}

// ---------------------------------------------------------------------------
// WMMA bf16 GEMM: C[M,N] = A[M,K](bf16,row-major,lda) * Bt[N,K](bf16) (+bias)(+gelu)
// Block = 8 waves = 128 rows of A x (NBLK*16) cols, staged through LDS in
// K-chunks of 128 via GLOBAL_LOAD_ASYNC_TO_LDS_B128 (ASYNCcnt). Each wave
// computes a 16 x (NBLK*16) stripe: one A fragment feeds NBLK back-to-back
// v_wmma_f32_16x16x32_bf16 accumulations.
// ---------------------------------------------------------------------------
template <int NBLK>
__global__ __launch_bounds__(256)
void k_gemm_bf16(const unsigned short* __restrict__ A, int lda,
                 const unsigned short* __restrict__ Bt, int K,
                 float* __restrict__ Cm, int ldc,
                 int M, int N,
                 const float* __restrict__ bias, int act)
{
    __shared__ unsigned short Alds[128 * 128];
    __shared__ unsigned short Blds[NBLK * 16 * 128];

    const int tid   = threadIdx.x;
    const int wv    = tid >> 5;
    const int lane  = tid & 31;
    const int mTiles  = M >> 4;
    const int nGroups = N / (16 * NBLK);
    const int mb = (int)blockIdx.x / nGroups;
    const int ng = (int)blockIdx.x % nGroups;
    const int rowBase = mb * 128;
    const int colBase = ng * (16 * NBLK);
    const int myMTile = (mb << 3) + wv;
    const bool active = (myMTile < mTiles);

    v8f acc[NBLK] = {};
    union Frag { v16bf v; uint4 u[2]; };

    for (int kc = 0; kc < K; kc += 128) {
        const int chunk = (K - kc < 128) ? (K - kc) : 128;
        const int gpr = chunk >> 3;                 // 16B granules per row
        // stage A slab (128 x chunk)
        for (int g = tid; g < 128 * gpr; g += 256) {
            int row = g / gpr;
            int ko  = (g - row * gpr) << 3;
            int grow = rowBase + row;
            if (grow < M) {
                const unsigned short* gp = A + (size_t)grow * lda + kc + ko;
                stage16(gp, Alds + row * chunk + ko);
                if (kc + 128 < K) __builtin_prefetch(gp + 128, 0, 1);
            }
        }
        // stage B slab (NBLK*16 x chunk)
        for (int g = tid; g < NBLK * 16 * gpr; g += 256) {
            int row = g / gpr;
            int ko  = (g - row * gpr) << 3;
            const unsigned short* gp = Bt + (size_t)(colBase + row) * K + kc + ko;
            stage16(gp, Blds + row * chunk + ko);
        }
#if HAVE_ASYNC_LDS
        __builtin_amdgcn_s_wait_asynccnt(0);
#endif
        __syncthreads();
        if (active) {
            const unsigned short* arow =
                Alds + (size_t)((wv << 4) + (lane & 15)) * chunk + ((lane >> 4) << 3);
            for (int k0 = 0; k0 < chunk; k0 += 32) {
                Frag fa;
                fa.u[0] = *(const uint4*)(arow + k0);        // K = base+0..7
                fa.u[1] = *(const uint4*)(arow + k0 + 16);   // K = base+16..23
#pragma unroll
                for (int nb = 0; nb < NBLK; ++nb) {
                    Frag fb;
                    const unsigned short* brow =
                        Blds + (size_t)((nb << 4) + (lane & 15)) * chunk + k0 + ((lane >> 4) << 3);
                    fb.u[0] = *(const uint4*)(brow);         // K = base+0..7
                    fb.u[1] = *(const uint4*)(brow + 16);    // K = base+16..23
                    acc[nb] = __builtin_amdgcn_wmma_f32_16x16x32_bf16(
                                  false, fa.v, false, fb.v, (short)0, acc[nb], false, false);
                }
            }
        }
        __syncthreads();
    }
    if (!active) return;

    int crow = (myMTile << 4) + ((lane >> 4) << 3);
#pragma unroll
    for (int nb = 0; nb < NBLK; ++nb) {
        int ccol = colBase + (nb << 4) + (lane & 15);
        float bb = bias ? bias[ccol] : 0.0f;
        float* cp = Cm + (size_t)crow * ldc + ccol;
#pragma unroll
        for (int j = 0; j < 8; ++j) {
            float v = acc[nb][j] + bb;
            if (act == 1) v = geluf(v);
            cp[(size_t)j * ldc] = v;
        }
    }
}

// --------------------------- small utility kernels -------------------------
__global__ void k_fill(float* p, int n, float v) {
    int i = blockIdx.x * 256 + threadIdx.x;
    if (i < n) p[i] = v;
}
__global__ void k_conv_bf16(const float* __restrict__ s, unsigned short* __restrict__ d,
                            int n, int dogelu) {
    int i = blockIdx.x * 256 + threadIdx.x;
    if (i >= n) return;
    float v = s[i];
    if (dogelu) v = geluf(v);
    d[i] = f2bf(v);
}
// dst[b][n][k] = bf16(src[b][k][n])   (weight transpose + convert)
__global__ void k_wt(const float* __restrict__ s, unsigned short* __restrict__ d,
                     int B, int K, int N) {
    int i = blockIdx.x * 256 + threadIdx.x;
    if (i >= B * K * N) return;
    int b = i / (K * N);
    int r = i - b * K * N;
    int n = r / K;
    int k = r - n * K;
    d[i] = f2bf(s[(size_t)b * K * N + (size_t)k * N + n]);
}
__global__ void k_embed(const float* __restrict__ x, const float* __restrict__ W,
                        const float* __restrict__ b, float* __restrict__ out) {
    int i = blockIdx.x * 256 + threadIdx.x;
    if (i >= C_NODES * 256) return;
    int c = i >> 8, j = i & 255;
    float s = b[j];
    const float* xr = x + c * 16;
#pragma unroll
    for (int k = 0; k < 16; ++k) s += xr[k] * W[k * 256 + j];
    out[i] = geluf(s);
}
__global__ void k_rmsnorm(const float* __restrict__ x, const float* __restrict__ scale,
                          float* __restrict__ out, int rows) {
    int r = blockIdx.x, t = threadIdx.x;
    __shared__ float red[256];
    float v = x[(size_t)r * 256 + t];
    red[t] = v * v; __syncthreads();
    for (int s = 128; s > 0; s >>= 1) { if (t < s) red[t] += red[t + s]; __syncthreads(); }
    float ms = red[0] * (1.0f / 256.0f) + 1e-6f;
    out[(size_t)r * 256 + t] = scale[t] * v * rsqrtf(ms);
}
// out[j] = act( sum_k vec[k]*W[k*N+j] + b[j] )
__global__ void k_vecmat(const float* __restrict__ vec, const float* __restrict__ W,
                         const float* __restrict__ b, float* __restrict__ out,
                         int K, int N, int act) {
    int j = blockIdx.x * 256 + threadIdx.x;
    if (j >= N) return;
    float s = b[j];
    for (int k = 0; k < K; ++k) s += vec[k] * W[(size_t)k * N + j];
    out[j] = act ? geluf(s) : s;
}

// ------------------------------- GAT kernels -------------------------------
__global__ void k_gat_dots(const float* __restrict__ xh, const float* __restrict__ attS,
                           const float* __restrict__ attD,
                           float* __restrict__ as_, float* __restrict__ ad_) {
    int i = blockIdx.x * 256 + threadIdx.x;
    if (i >= C_NODES * 4) return;
    int c = i >> 2, h = i & 3;
    const float* xr = xh + (size_t)c * 256 + h * 64;
    float s = 0.f, d = 0.f;
#pragma unroll 8
    for (int t = 0; t < 64; ++t) { s += xr[t] * attS[h * 64 + t]; d += xr[t] * attD[h * 64 + t]; }
    as_[i] = s; ad_[i] = d;
}
__global__ void k_gat_logit_max(float* __restrict__ elog, const float* __restrict__ as_,
                                const float* __restrict__ ad_, const int* __restrict__ ei,
                                float* __restrict__ mx) {
    int i = blockIdx.x * 256 + threadIdx.x;
    if (i >= EWL * 4) return;
    int e = i >> 2, h = i & 3;
    int s, d;
    if (e < E_EDGES) { s = ei[e]; d = ei[E_EDGES + e]; } else { s = d = e - E_EDGES; }
    float lg = leakyf(as_[s * 4 + h] + ad_[d * 4 + h]);
    elog[i] = lg;
    atomicMaxF(&mx[d * 4 + h], lg);
}
__global__ void k_gat_exp_sum(float* __restrict__ elog, const int* __restrict__ ei,
                              const float* __restrict__ mx, float* __restrict__ sm) {
    int i = blockIdx.x * 256 + threadIdx.x;
    if (i >= EWL * 4) return;
    int e = i >> 2, h = i & 3;
    int d = (e < E_EDGES) ? ei[E_EDGES + e] : (e - E_EDGES);
    float ex = __expf(elog[i] - mx[d * 4 + h]);
    elog[i] = ex;
    atomicAdd(&sm[d * 4 + h], ex);
}
__global__ void k_gat_scatter(const float* __restrict__ elog, const int* __restrict__ ei,
                              const float* __restrict__ sm, const float* __restrict__ xh,
                              float* __restrict__ acc) {
    int i = blockIdx.x * 256 + threadIdx.x;
    if (i >= EWL * 64) return;
    int e = i >> 6, q = i & 63;
    int s, d;
    if (e < E_EDGES) { s = ei[e]; d = ei[E_EDGES + e]; } else { s = d = e - E_EDGES; }
    int h = q >> 4, d0 = q << 2;
    float a = elog[e * 4 + h] / (sm[d * 4 + h] + 1e-16f);
    const float* xr = xh + (size_t)s * 256 + d0;
    float* ar = acc + (size_t)d * 256 + d0;
#pragma unroll
    for (int t = 0; t < 4; ++t) atomicAdd(&ar[t], a * xr[t]);
}
__global__ void k_gat_combine(const float* __restrict__ xin, const float* __restrict__ acc,
                              const float* __restrict__ b0, const float* __restrict__ b1,
                              const float* __restrict__ b2, float* __restrict__ xout) {
    int i = blockIdx.x * 256 + threadIdx.x;
    if (i >= C_NODES * 256) return;
    int d = i & 255;
    xout[i] = leakyf(xin[i] + acc[i] + b0[d] + b1[d] + b2[d]);
}

// ------------------------------- HGT kernels -------------------------------
__global__ void k_gather_bf16(const float* __restrict__ feat, const int* __restrict__ idx,
                              unsigned short* __restrict__ dst) {
    int i = blockIdx.x * 256 + threadIdx.x;
    if (i >= E_EDGES * 256) return;
    int e = i >> 8, j = i & 255;
    dst[i] = f2bf(feat[(size_t)idx[e] * 256 + j]);
}
__global__ void k_hgt_logit(const float* __restrict__ krel, const float* __restrict__ q,
                            const int* __restrict__ dst, const float* __restrict__ prel,
                            int r, float* __restrict__ alog, float* __restrict__ mx) {
    int i = blockIdx.x * 256 + threadIdx.x;
    if (i >= E_EDGES * 8) return;
    int e = i >> 3, h = i & 7;
    int d = dst[e];
    const float* qr = q + (size_t)d * 256 + h * 32;
    const float* kr = krel + (size_t)e * 256 + h * 32;
    float s = 0.f;
#pragma unroll 8
    for (int t = 0; t < 32; ++t) s += qr[t] * kr[t];
    s *= prel[r * 8 + h] * 0.17677669529663689f;  // 1/sqrt(32)
    alog[((size_t)r * E_EDGES + e) * 8 + h] = s;
    atomicMaxF(&mx[d * 8 + h], s);
}
__global__ void k_hgt_exp_sum(float* __restrict__ alog, const int* __restrict__ d0,
                              const int* __restrict__ d1, const int* __restrict__ d2,
                              const float* __restrict__ mx, float* __restrict__ sm) {
    size_t i = (size_t)blockIdx.x * 256 + threadIdx.x;
    if (i >= (size_t)3 * E_EDGES * 8) return;
    int g = (int)(i >> 3), h = (int)(i & 7);
    int r = g / E_EDGES, e = g - r * E_EDGES;
    int d = (r == 0 ? d0 : (r == 1 ? d1 : d2))[e];
    float ex = __expf(alog[i] - mx[d * 8 + h]);
    alog[i] = ex;
    atomicAdd(&sm[d * 8 + h], ex);
}
__global__ void k_hgt_scatter(const float* __restrict__ alog, const int* __restrict__ d0,
                              const int* __restrict__ d1, const int* __restrict__ d2,
                              const float* __restrict__ sm, const float* __restrict__ msg,
                              float* __restrict__ hout) {
    size_t i = (size_t)blockIdx.x * 256 + threadIdx.x;
    if (i >= (size_t)3 * E_EDGES * 64) return;
    int g = (int)(i >> 6), q = (int)(i & 63);
    int r = g / E_EDGES, e = g - r * E_EDGES;
    int d = (r == 0 ? d0 : (r == 1 ? d1 : d2))[e];
    int h = q >> 3, c0 = q << 2;
    float w = alog[(size_t)g * 8 + h] / (sm[d * 8 + h] + 1e-16f);
    const float* mr = msg + (size_t)g * 256 + c0;
    float* hr = hout + (size_t)d * 256 + c0;
#pragma unroll
    for (int t = 0; t < 4; ++t) atomicAdd(&hr[t], w * mr[t]);
}
__global__ void k_hgt_combine(const float* __restrict__ x2, const float* __restrict__ houtA,
                              const float* __restrict__ inj, const float* __restrict__ fp,
                              const float* __restrict__ skip, float* __restrict__ nodes) {
    int i = blockIdx.x * 256 + threadIdx.x;
    if (i >= C_NODES * 256) return;
    int d = i & 255;
    float g = 1.0f / (1.0f + __expf(-skip[0]));
    float hg = g * houtA[i] + (1.0f - g) * x2[i];
    float x3 = leakyf(x2[i] + hg);
    nodes[i] = fp[d] * (x3 + inj[i]) + fp[256 + d];
}

// ----------------------------- h-module kernels ----------------------------
__global__ void k_hpool_logits(const float* __restrict__ Qh, const float* __restrict__ Kb,
                               float* __restrict__ wl) {
    int i = blockIdx.x * 256 + threadIdx.x;
    if (i >= C_NODES * 4) return;
    int c = i >> 2, hp = i & 3;
    const float* qr = Qh + hp * 64;
    const float* kr = Kb + (size_t)c * 256 + hp * 64;
    float s = 0.f;
#pragma unroll 8
    for (int t = 0; t < 64; ++t) s += qr[t] * kr[t];
    wl[(size_t)hp * C_NODES + c] = s * 0.125f;   // 1/sqrt(64)
}
__global__ void k_softmax_big(float* __restrict__ w) {
    int hp = blockIdx.x, t = threadIdx.x;
    float* row = w + (size_t)hp * C_NODES;
    __shared__ float red[256];
    __shared__ float smax, ssum;
    float m = -1e30f;
    for (int i = t; i < C_NODES; i += 256) m = fmaxf(m, row[i]);
    red[t] = m; __syncthreads();
    for (int s = 128; s > 0; s >>= 1) { if (t < s) red[t] = fmaxf(red[t], red[t + s]); __syncthreads(); }
    if (t == 0) smax = red[0];
    __syncthreads();
    float su = 0.f;
    for (int i = t; i < C_NODES; i += 256) su += __expf(row[i] - smax);
    red[t] = su; __syncthreads();
    for (int s = 128; s > 0; s >>= 1) { if (t < s) red[t] += red[t + s]; __syncthreads(); }
    if (t == 0) ssum = red[0];
    __syncthreads();
    for (int i = t; i < C_NODES; i += 256) row[i] = __expf(row[i] - smax) / ssum;
}
__global__ void k_hpool_weighted(const float* __restrict__ w, const float* __restrict__ Vb,
                                 float* __restrict__ pooled) {
    int j = blockIdx.x, t = threadIdx.x;   // j in [0,256)
    int hp = j >> 6;
    __shared__ float red[256];
    float s = 0.f;
    for (int c = t; c < C_NODES; c += 256)
        s += w[(size_t)hp * C_NODES + c] * Vb[(size_t)c * 256 + j];
    red[t] = s; __syncthreads();
    for (int k = 128; k > 0; k >>= 1) { if (t < k) red[t] += red[t + k]; __syncthreads(); }
    if (t == 0) pooled[j] = red[0];
}
__global__ void k_concat_zc(const float* __restrict__ pooled, const float* __restrict__ z,
                            float* __restrict__ zc) {
    int j = threadIdx.x + blockIdx.x * 256;
    if (j >= 512) return;
    zc[j] = (j < 256) ? pooled[j] : z[j - 256];
}

// ------------------------------- readout -----------------------------------
__global__ void k_build_cat(const float* __restrict__ nodes, const float* __restrict__ z,
                            unsigned short* __restrict__ cat) {
    size_t i = (size_t)blockIdx.x * 256 + threadIdx.x;
    if (i >= (size_t)C_NODES * 512) return;
    int c = (int)(i >> 9), j = (int)(i & 511);
    float v = (j < 256) ? nodes[(size_t)c * 256 + j] : z[j - 256];
    cat[i] = f2bf(v);
}
__global__ void k_ro2(const float* __restrict__ r1, const float* __restrict__ W2,
                      const float* __restrict__ b2, float* __restrict__ out) {
    int c = blockIdx.x * 256 + threadIdx.x;
    if (c >= C_NODES) return;
    const float* rr = r1 + (size_t)c * 256;
    float s = b2[0];
    for (int j = 0; j < 256; ++j) s += rr[j] * W2[j];
    out[c] = s;
}

// ===========================================================================
extern "C" void kernel_launch(void* const* d_in, const int* in_sizes, int n_in,
                              void* d_out, int out_size, void* d_ws, size_t ws_size,
                              hipStream_t stream) {
    (void)in_sizes; (void)n_in; (void)out_size; (void)ws_size;
    // ---- input mapping: x_cell, params(tree-flattened, sorted keys), ei_line, ei_region, ei_diag
    const float* x_cell   = (const float*)d_in[0];
    const float* embW     = (const float*)d_in[1];
    const float* embB     = (const float*)d_in[2];
    const float* embScale = (const float*)d_in[3];
    const float* filmW1   = (const float*)d_in[4];
    const float* filmW2   = (const float*)d_in[5];
    const float* filmB1   = (const float*)d_in[6];
    const float* filmB2   = (const float*)d_in[7];
    const float *gatW[6], *gatAd[6], *gatAs[6], *gatB[6];
    for (int g = 0; g < 6; ++g) {
        int base = 8 + g * 4;
        gatW[g]  = (const float*)d_in[base + 0];
        gatAd[g] = (const float*)d_in[base + 1];
        gatAs[g] = (const float*)d_in[base + 2];
        gatB[g]  = (const float*)d_in[base + 3];
    }
    const float* hW1 = (const float*)d_in[32];
    const float* hW2 = (const float*)d_in[33];
    const float* hWk = (const float*)d_in[34];
    const float* hWq = (const float*)d_in[35];
    const float* hWv = (const float*)d_in[36];
    const float* hb1 = (const float*)d_in[37];
    const float* hb2 = (const float*)d_in[38];
    const float* hbk = (const float*)d_in[39];
    const float* hbq = (const float*)d_in[40];
    const float* hbv = (const float*)d_in[41];
    const float* hscale = (const float*)d_in[42];
    const float* hgtWa = (const float*)d_in[43];
    const float* hgtWk = (const float*)d_in[44];
    const float* hgtWq = (const float*)d_in[45];
    const float* hgtWv = (const float*)d_in[46];
    const float* a_rel = (const float*)d_in[47];
    const float* hgtBa = (const float*)d_in[48];
    const float* hgtBk = (const float*)d_in[49];
    const float* hgtBq = (const float*)d_in[50];
    const float* hgtBv = (const float*)d_in[51];
    const float* m_rel = (const float*)d_in[52];
    const float* p_rel = (const float*)d_in[53];
    const float* skipP = (const float*)d_in[54];
    const float* injW  = (const float*)d_in[55];
    const float* injB  = (const float*)d_in[56];
    const float* roW1  = (const float*)d_in[57];
    const float* roW2  = (const float*)d_in[58];
    const float* roB1  = (const float*)d_in[59];
    const float* roB2  = (const float*)d_in[60];
    const int* eis[3] = { (const int*)d_in[61], (const int*)d_in[62], (const int*)d_in[63] };

    // ---- workspace carve-out
    char* wsb = (char*)d_ws;
    size_t off = 0;
    auto alloc = [&](size_t bytes) -> void* {
        void* p = (void*)(wsb + off);
        off = (off + bytes + 255) & ~(size_t)255;
        return p;
    };
    const size_t ND = (size_t)C_NODES * 256;
    unsigned short* gatWT[6];
    for (int g = 0; g < 6; ++g) gatWT[g] = (unsigned short*)alloc(65536 * 2);
    unsigned short* hgtWkT = (unsigned short*)alloc(65536 * 2);
    unsigned short* hgtWqT = (unsigned short*)alloc(65536 * 2);
    unsigned short* hgtWvT = (unsigned short*)alloc(65536 * 2);
    unsigned short* hgtWaT = (unsigned short*)alloc(65536 * 2);
    unsigned short* arelT  = (unsigned short*)alloc(24576 * 2);
    unsigned short* mrelT  = (unsigned short*)alloc(24576 * 2);
    unsigned short* hWkT   = (unsigned short*)alloc(65536 * 2);
    unsigned short* hWvT   = (unsigned short*)alloc(65536 * 2);
    unsigned short* injWT  = (unsigned short*)alloc(65536 * 2);
    unsigned short* roW1T  = (unsigned short*)alloc(131072 * 2);
    unsigned short* nbf    = (unsigned short*)alloc(ND * 2);
    unsigned short* kgbf   = (unsigned short*)alloc((size_t)E_EDGES * 256 * 2);
    unsigned short* vgbf   = (unsigned short*)alloc((size_t)E_EDGES * 256 * 2);
    unsigned short* catbf  = (unsigned short*)alloc((size_t)C_NODES * 512 * 2);
    float* nodes  = (float*)alloc(ND * 4);
    float* x1     = (float*)alloc(ND * 4);
    float* x2     = (float*)alloc(ND * 4);
    float* inj    = (float*)alloc(ND * 4);
    float* xh     = (float*)alloc(ND * 4);
    float* gatacc = (float*)alloc(ND * 4);
    float* houtA  = (float*)alloc(ND * 4);
    float* hout   = (float*)alloc(ND * 4);
    float* kbuf   = (float*)alloc(ND * 4);
    float* qbuf   = (float*)alloc(ND * 4);
    float* vbuf   = (float*)alloc(ND * 4);
    float* Kb     = (float*)alloc(ND * 4);
    float* Vb     = (float*)alloc(ND * 4);
    float* r1     = (float*)alloc(ND * 4);
    float* as_    = (float*)alloc((size_t)C_NODES * 4 * 4);
    float* ad_    = (float*)alloc((size_t)C_NODES * 4 * 4);
    float* mxb    = (float*)alloc((size_t)C_NODES * 8 * 4);
    float* smb    = (float*)alloc((size_t)C_NODES * 8 * 4);
    float* elog   = (float*)alloc((size_t)EWL * 4 * 4);
    float* krel   = (float*)alloc((size_t)E_EDGES * 256 * 4);
    float* msg    = (float*)alloc((size_t)3 * E_EDGES * 256 * 4);
    float* alog   = (float*)alloc((size_t)3 * E_EDGES * 8 * 4);
    float* fh1    = (float*)alloc(512 * 4);
    float* fp     = (float*)alloc(512 * 4);
    float* zvec   = (float*)alloc(256 * 4);
    float* Qh     = (float*)alloc(256 * 4);
    float* wlog   = (float*)alloc((size_t)4 * C_NODES * 4);
    float* pooled = (float*)alloc(256 * 4);
    float* zc     = (float*)alloc(512 * 4);
    float* h1v    = (float*)alloc(256 * 4);
    float* h2v    = (float*)alloc(256 * 4);

#define LAU(kern, n, ...) kern<<<dim3((unsigned)(((size_t)(n) + 255) / 256)), dim3(256), 0, stream>>>(__VA_ARGS__)
    auto gemm = [&](const unsigned short* A, int lda, const unsigned short* Bt, int K,
                    float* Cm, int ldc, int M, int N, const float* bias, int act) {
        int mBlocks = ((M >> 4) + 7) >> 3;
        if ((N & 63) == 0) {
            k_gemm_bf16<4><<<dim3((unsigned)(mBlocks * (N >> 6))), dim3(256), 0, stream>>>(
                A, lda, Bt, K, Cm, ldc, M, N, bias, act);
        } else {
            k_gemm_bf16<2><<<dim3((unsigned)(mBlocks * (N >> 5))), dim3(256), 0, stream>>>(
                A, lda, Bt, K, Cm, ldc, M, N, bias, act);
        }
    };

    // ---- weight prep (transpose+bf16) ----
    for (int g = 0; g < 6; ++g) LAU(k_wt, 65536, gatW[g], gatWT[g], 1, 256, 256);
    LAU(k_wt, 65536, hgtWk, hgtWkT, 1, 256, 256);
    LAU(k_wt, 65536, hgtWq, hgtWqT, 1, 256, 256);
    LAU(k_wt, 65536, hgtWv, hgtWvT, 1, 256, 256);
    LAU(k_wt, 65536, hgtWa, hgtWaT, 1, 256, 256);
    LAU(k_wt, 24576, a_rel, arelT, 24, 32, 32);
    LAU(k_wt, 24576, m_rel, mrelT, 24, 32, 32);
    LAU(k_wt, 65536, hWk, hWkT, 1, 256, 256);
    LAU(k_wt, 65536, hWv, hWvT, 1, 256, 256);
    LAU(k_wt, 65536, injW, injWT, 1, 256, 256);
    LAU(k_wt, 131072, roW1, roW1T, 1, 512, 256);

    // ---- embed -> nodes, inj, z=0 ----
    LAU(k_embed, ND, x_cell, embW, embB, xh);
    k_rmsnorm<<<dim3(C_NODES), dim3(256), 0, stream>>>(xh, embScale, nodes, C_NODES);
    LAU(k_conv_bf16, ND, nodes, nbf, (int)ND, 0);
    gemm(nbf, 256, injWT, 256, inj, 256, C_NODES, 256, injB, 0);
    LAU(k_fill, 256, zvec, 256, 0.0f);

    auto gat_layer = [&](const float* xin, float* xout, int layer) {
        LAU(k_conv_bf16, ND, xin, nbf, (int)ND, 0);
        LAU(k_fill, ND, gatacc, (int)ND, 0.0f);
        for (int i = 0; i < 3; ++i) {
            int g = layer * 3 + i;
            gemm(nbf, 256, gatWT[g], 256, xh, 256, C_NODES, 256, nullptr, 0);
            LAU(k_gat_dots, C_NODES * 4, xh, gatAs[g], gatAd[g], as_, ad_);
            LAU(k_fill, C_NODES * 4, mxb, C_NODES * 4, -1e30f);
            LAU(k_fill, C_NODES * 4, smb, C_NODES * 4, 0.0f);
            LAU(k_gat_logit_max, EWL * 4, elog, as_, ad_, eis[i], mxb);
            LAU(k_gat_exp_sum, EWL * 4, elog, eis[i], mxb, smb);
            LAU(k_gat_scatter, (size_t)EWL * 64, elog, eis[i], smb, xh, gatacc);
        }
        LAU(k_gat_combine, ND, xin, gatacc, gatB[layer * 3], gatB[layer * 3 + 1],
            gatB[layer * 3 + 2], xout);
    };

    for (int cyc = 0; cyc < 2; ++cyc) {
        for (int mic = 0; mic < 2; ++mic) {
            // FiLM from z
            LAU(k_vecmat, 512, zvec, filmW1, filmB1, fh1, 256, 512, 1);
            LAU(k_vecmat, 512, fh1, filmW2, filmB2, fp, 512, 512, 0);
            // two GAT layers
            gat_layer(nodes, x1, 0);
            gat_layer(x1, x2, 1);
            // HGT
            LAU(k_conv_bf16, ND, x2, nbf, (int)ND, 0);
            gemm(nbf, 256, hgtWkT, 256, kbuf, 256, C_NODES, 256, hgtBk, 0);
            gemm(nbf, 256, hgtWqT, 256, qbuf, 256, C_NODES, 256, hgtBq, 0);
            gemm(nbf, 256, hgtWvT, 256, vbuf, 256, C_NODES, 256, hgtBv, 0);
            LAU(k_fill, C_NODES * 8, mxb, C_NODES * 8, -1e30f);
            LAU(k_fill, C_NODES * 8, smb, C_NODES * 8, 0.0f);
            LAU(k_fill, ND, hout, (int)ND, 0.0f);
            for (int r = 0; r < 3; ++r) {
                const int* src = eis[r];
                const int* dst = eis[r] + E_EDGES;
                LAU(k_gather_bf16, (size_t)E_EDGES * 256, kbuf, src, kgbf);
                LAU(k_gather_bf16, (size_t)E_EDGES * 256, vbuf, src, vgbf);
                for (int h = 0; h < 8; ++h) {
                    gemm(kgbf + h * 32, 256, arelT + (size_t)(r * 8 + h) * 1024, 32,
                         krel + h * 32, 256, E_EDGES, 32, nullptr, 0);
                    gemm(vgbf + h * 32, 256, mrelT + (size_t)(r * 8 + h) * 1024, 32,
                         msg + (size_t)r * E_EDGES * 256 + h * 32, 256, E_EDGES, 32, nullptr, 0);
                }
                LAU(k_hgt_logit, E_EDGES * 8, krel, qbuf, dst, p_rel, r, alog, mxb);
            }
            LAU(k_hgt_exp_sum, (size_t)3 * E_EDGES * 8, alog, eis[0] + E_EDGES,
                eis[1] + E_EDGES, eis[2] + E_EDGES, mxb, smb);
            LAU(k_hgt_scatter, (size_t)3 * E_EDGES * 64, alog, eis[0] + E_EDGES,
                eis[1] + E_EDGES, eis[2] + E_EDGES, smb, msg, hout);
            LAU(k_conv_bf16, ND, hout, nbf, (int)ND, 1);               // gelu + bf16
            gemm(nbf, 256, hgtWaT, 256, houtA, 256, C_NODES, 256, hgtBa, 0);
            LAU(k_hgt_combine, ND, x2, houtA, inj, fp, skipP, nodes);
        }
        // ---- h_module: update z ----
        LAU(k_vecmat, 256, zvec, hWq, hbq, Qh, 256, 256, 0);
        LAU(k_conv_bf16, ND, nodes, nbf, (int)ND, 0);
        gemm(nbf, 256, hWkT, 256, Kb, 256, C_NODES, 256, hbk, 0);
        gemm(nbf, 256, hWvT, 256, Vb, 256, C_NODES, 256, hbv, 0);
        LAU(k_hpool_logits, C_NODES * 4, Qh, Kb, wlog);
        k_softmax_big<<<dim3(4), dim3(256), 0, stream>>>(wlog);
        k_hpool_weighted<<<dim3(256), dim3(256), 0, stream>>>(wlog, Vb, pooled);
        LAU(k_concat_zc, 512, pooled, zvec, zc);
        LAU(k_vecmat, 256, zc, hW1, hb1, h1v, 512, 256, 1);
        LAU(k_vecmat, 256, h1v, hW2, hb2, h2v, 256, 256, 0);
        k_rmsnorm<<<dim3(1), dim3(256), 0, stream>>>(h2v, hscale, zvec, 1);
    }

    // ---- readout ----
    LAU(k_build_cat, (size_t)C_NODES * 512, nodes, zvec, catbf);
    gemm(catbf, 512, roW1T, 512, r1, 256, C_NODES, 256, roB1, 1);
    LAU(k_ro2, C_NODES, r1, roW2, roB2, (float*)d_out);
#undef LAU
}
